// BNN1D_14448269984213
// MI455X (gfx1250) — 8-bit matrix operand order repaired and checked by reading, compiled, not run on hardware
//
#include <hip/hip_runtime.h>

typedef int v8i __attribute__((ext_vector_type(8)));
typedef int gv4i __attribute__((vector_size(16)));

#define BN_EPS 1e-5f

#if defined(__has_builtin)
#if __has_builtin(__builtin_amdgcn_global_load_async_to_lds_b128) && \
    __has_builtin(__builtin_amdgcn_s_wait_asynccnt)
#define HAVE_ASYNC_LDS 1
#endif
#endif

__device__ __forceinline__ void copy_g2l_b128(signed char* lds_dst,
                                              const signed char* g_src) {
#if defined(HAVE_ASYNC_LDS)
  __builtin_amdgcn_global_load_async_to_lds_b128(
      (__attribute__((address_space(1))) gv4i*)(g_src),
      (__attribute__((address_space(3))) gv4i*)(lds_dst), 0, 0);
#else
  *(int4*)lds_dst = *(const int4*)g_src;
#endif
}

__device__ __forceinline__ void wait_async_copies() {
#if defined(HAVE_ASYNC_LDS)
  __builtin_amdgcn_s_wait_asynccnt(0);
#endif
}

__global__ __launch_bounds__(256) void bnn_prep_kernel(
    const float* __restrict__ w2, const float* __restrict__ w3,
    const float* __restrict__ wf, float* __restrict__ scal,
    float* __restrict__ h3acc, signed char* __restrict__ w2s,
    signed char* __restrict__ w3s) {
  __shared__ float red[256];
  const int t = threadIdx.x;
  for (int i = t; i < 512 * 128; i += 256) h3acc[i] = 0.0f;
  for (int i = t; i < 64 * 192; i += 256) {
    const int kk = i % 192;
    w2s[i] = (kk < 160) ? ((w2[(i / 192) * 160 + kk] >= 0.0f) ? 1 : -1) : 0;
  }
  for (int i = t; i < 128 * 192; i += 256)
    w3s[i] = (w3[i] >= 0.0f) ? 1 : -1;
  const float* src[3] = {w2, w3, wf};
  const int n[3] = {64 * 32 * 5, 128 * 64 * 3, 64 * 128};
  for (int j = 0; j < 3; ++j) {
    float m = 0.0f;
    for (int i = t; i < n[j]; i += 256) m = fmaxf(m, fabsf(src[j][i]));
    red[t] = m;
    __syncthreads();
    for (int s = 128; s > 0; s >>= 1) {
      if (t < s) red[t] = fmaxf(red[t], red[t + s]);
      __syncthreads();
    }
    if (t == 0) scal[j] = red[0];
    __syncthreads();
  }
}

__global__ __launch_bounds__(256) void bnn_first_conv_kernel(
    const float* __restrict__ x, const float* __restrict__ w1,
    const float* __restrict__ b1, const float* __restrict__ g1,
    const float* __restrict__ be1, const float* __restrict__ m1,
    const float* __restrict__ v1, _Float16* __restrict__ h1f) {
  const long idx = (long)blockIdx.x * blockDim.x + threadIdx.x;
  const long total = 512L * 32 * 2048;
  if (idx >= total) return;
  const int lp = (int)(idx % 2048);
  const int ch = (int)((idx / 2048) % 32);
  const int b  = (int)(idx / (2048L * 32));
  const float* xb = x + (long)b * 4096;
  float wv[5];
#pragma unroll
  for (int k = 0; k < 5; ++k) wv[k] = w1[ch * 5 + k];
  const float bnscale = g1[ch] * rsqrtf(v1[ch] + BN_EPS);
  float best = -1e30f;
#pragma unroll
  for (int j = 0; j < 2; ++j) {
    const int l = 2 * lp + j;
    float acc = b1[ch];
#pragma unroll
    for (int k = 0; k < 5; ++k) {
      const int p = l + k - 2;
      if (p >= 0 && p < 4096) acc += wv[k] * xb[p];
    }
    float y = (acc - m1[ch]) * bnscale + be1[ch];
    best = fmaxf(best, fmaxf(y, 0.0f));
  }
  h1f[idx] = (_Float16)best;
}

__global__ __launch_bounds__(128) void bnn_bconv1_kernel(
    const _Float16* __restrict__ h1f, const signed char* __restrict__ w2s,
    const float* __restrict__ b2, const float* __restrict__ g2,
    const float* __restrict__ be2, const float* __restrict__ m2,
    const float* __restrict__ v2, const float* __restrict__ sa1p,
    const float* __restrict__ scal, _Float16* __restrict__ h2f) {
  __shared__ __attribute__((aligned(16))) signed char Al[64 * 192];
  __shared__ __attribute__((aligned(16))) signed char Bl[16 * 192];
  const int t = threadIdx.x;
  const int sup = blockIdx.x & 15;
  const int b   = blockIdx.x >> 4;

#pragma unroll
  for (int it = 0; it < 6; ++it) {
    const int off = it * 2048 + t * 16;
    copy_g2l_b128(Al + off, w2s + off);
  }
  wait_async_copies();

  const int wv = t >> 5, lane = t & 31;
  const int half = lane >> 4, nl = lane & 15;
  const int mrow = wv * 16 + nl;
  const float scale = sa1p[0] * scal[0];
  float Ar[8], Cr[8];
#pragma unroll
  for (int r = 0; r < 8; ++r) {
    const int ch = wv * 16 + r + half * 8;
    const float bn = g2[ch] * rsqrtf(v2[ch] + BN_EPS);
    Ar[r] = scale * bn;
    Cr[r] = (b2[ch] - m2[ch]) * bn + be2[ch];
  }

  for (int nt = 0; nt < 8; ++nt) {
    const int tile = sup * 8 + nt;
    __syncthreads();
    for (int i = t; i < 16 * 192; i += 128) {
      const int n = i / 192, kk = i % 192;
      signed char v = 0;
      if (kk < 160) {
        const int c = kk / 5, k = kk % 5;
        const int p = tile * 16 + n + k - 2;
        if (p >= 0 && p < 2048)
          v = ((float)h1f[((long)b * 32 + c) * 2048 + p] >= 0.0f) ? 1 : -1;
      }
      Bl[i] = v;
    }
    __syncthreads();

    v8i acc = {0, 0, 0, 0, 0, 0, 0, 0};
#pragma unroll
    for (int kc = 0; kc < 3; ++kc) {
      v8i a, bb;
#pragma unroll
      for (int v = 0; v < 8; ++v) {
        const int ka = (v & 3) * 4 + half * 16 +
                       (v >> 2) * 32 + kc * 64;
        a[v] = *(const int*)&Al[mrow * 192 + ka];
        const int kb = (v & 3) * 4 + half * 16 + (v >> 2) * 32 + kc * 64;
        bb[v] = *(const int*)&Bl[nl * 192 + kb];
      }
      acc = __builtin_amdgcn_wmma_i32_16x16x64_iu8(true, a, true, bb, acc,
                                                   false, false);
    }

#pragma unroll
    for (int r = 0; r < 8; ++r) {
      float y = fmaxf(Ar[r] * (float)acc[r] + Cr[r], 0.0f);
      const float y2 = __shfl_xor(y, 1, 32);
      if ((nl & 1) == 0) {
        const float p = fmaxf(y, y2);
        const int ch = wv * 16 + r + half * 8;
        h2f[((long)b * 64 + ch) * 1024 + tile * 8 + (nl >> 1)] = (_Float16)p;
      }
    }
  }
}

__global__ __launch_bounds__(256) void bnn_bconv2_kernel(
    const _Float16* __restrict__ h2f, const signed char* __restrict__ w3s,
    const float* __restrict__ b3, const float* __restrict__ g3,
    const float* __restrict__ be3, const float* __restrict__ m3,
    const float* __restrict__ v3, const float* __restrict__ sa2p,
    const float* __restrict__ scal, float* __restrict__ h3acc) {
  __shared__ __attribute__((aligned(16))) signed char Al[128 * 192];
  __shared__ __attribute__((aligned(16))) signed char Bl[16 * 192];
  const int t = threadIdx.x;
  const int sup = blockIdx.x & 7;
  const int b   = blockIdx.x >> 3;

#pragma unroll
  for (int it = 0; it < 6; ++it) {
    const int off = it * 4096 + t * 16;
    copy_g2l_b128(Al + off, w3s + off);
  }
  wait_async_copies();

  const int wv = t >> 5, lane = t & 31;
  const int half = lane >> 4, nl = lane & 15;
  const int mrow = wv * 16 + nl;
  const float scale = sa2p[0] * scal[1];
  float Ar[8], Cr[8], csum[8];
#pragma unroll
  for (int r = 0; r < 8; ++r) {
    const int ch = wv * 16 + r + half * 8;
    const float bn = g3[ch] * rsqrtf(v3[ch] + BN_EPS);
    Ar[r] = scale * bn;
    Cr[r] = (b3[ch] - m3[ch]) * bn + be3[ch];
    csum[r] = 0.0f;
  }

  for (int nt = 0; nt < 8; ++nt) {
    const int tile = sup * 8 + nt;
    __syncthreads();
    for (int i = t; i < 16 * 192; i += 256) {
      const int n = i / 192, kk = i % 192;
      const int c = kk / 3, k = kk % 3;
      const int p = tile * 16 + n + k - 1;
      signed char v = 0;
      if (p >= 0 && p < 1024)
        v = ((float)h2f[((long)b * 64 + c) * 1024 + p] >= 0.0f) ? 1 : -1;
      Bl[i] = v;
    }
    __syncthreads();

    v8i acc = {0, 0, 0, 0, 0, 0, 0, 0};
#pragma unroll
    for (int kc = 0; kc < 3; ++kc) {
      v8i a, bb;
#pragma unroll
      for (int v = 0; v < 8; ++v) {
        const int ka = (v & 3) * 4 + half * 16 +
                       (v >> 2) * 32 + kc * 64;
        a[v] = *(const int*)&Al[mrow * 192 + ka];
        const int kb = (v & 3) * 4 + half * 16 + (v >> 2) * 32 + kc * 64;
        bb[v] = *(const int*)&Bl[nl * 192 + kb];
      }
      acc = __builtin_amdgcn_wmma_i32_16x16x64_iu8(true, a, true, bb, acc,
                                                   false, false);
    }

#pragma unroll
    for (int r = 0; r < 8; ++r) {
      float y = fmaxf(Ar[r] * (float)acc[r] + Cr[r], 0.0f);
      y += __shfl_xor(y, 1, 32);
      y += __shfl_xor(y, 2, 32);
      y += __shfl_xor(y, 4, 32);
      y += __shfl_xor(y, 8, 32);
      csum[r] += y;
    }
  }

  if (nl == 0) {
#pragma unroll
    for (int r = 0; r < 8; ++r) {
      const int ch = wv * 16 + r + half * 8;
      atomicAdd(&h3acc[(long)b * 128 + ch], csum[r] * (1.0f / 1024.0f));
    }
  }
}

__global__ __launch_bounds__(64) void bnn_head_kernel(
    const float* __restrict__ h3acc, const float* __restrict__ wf,
    const float* __restrict__ bf, const float* __restrict__ g4,
    const float* __restrict__ be4, const float* __restrict__ m4,
    const float* __restrict__ v4, const float* __restrict__ wl,
    const float* __restrict__ bl, const float* __restrict__ sa3p,
    const float* __restrict__ scal, float* __restrict__ out) {
  __shared__ float hid[64];
  const int b = blockIdx.x, t = threadIdx.x;
  const float scale = sa3p[0] * scal[2];
  float acc = 0.0f;
  for (int c = 0; c < 128; ++c) {
    const float s = (h3acc[(long)b * 128 + c] >= 0.0f) ? 1.0f : -1.0f;
    const float w = (wf[t * 128 + c] >= 0.0f) ? 1.0f : -1.0f;
    acc += s * w;
  }
  float y = scale * acc + bf[t];
  y = (y - m4[t]) * (g4[t] * rsqrtf(v4[t] + BN_EPS)) + be4[t];
  hid[t] = fmaxf(y, 0.0f);
  __syncthreads();
  if (t < 10) {
    float o = bl[t];
#pragma unroll
    for (int c = 0; c < 64; ++c) o += wl[t * 64 + c] * hid[c];
    out[(long)b * 10 + t] = o;
  }
}

extern "C" void kernel_launch(void* const* d_in, const int* in_sizes, int n_in,
                              void* d_out, int out_size, void* d_ws,
                              size_t ws_size, hipStream_t stream) {
  (void)in_sizes; (void)n_in; (void)out_size; (void)ws_size;
  const float* x   = (const float*)d_in[0];
  const float* w1  = (const float*)d_in[1];
  const float* b1  = (const float*)d_in[2];
  const float* g1  = (const float*)d_in[3];
  const float* be1 = (const float*)d_in[4];
  const float* m1  = (const float*)d_in[5];
  const float* v1  = (const float*)d_in[6];
  const float* w2  = (const float*)d_in[7];
  const float* b2  = (const float*)d_in[8];
  const float* g2  = (const float*)d_in[9];
  const float* be2 = (const float*)d_in[10];
  const float* m2  = (const float*)d_in[11];
  const float* v2  = (const float*)d_in[12];
  const float* w3  = (const float*)d_in[13];
  const float* b3  = (const float*)d_in[14];
  const float* g3  = (const float*)d_in[15];
  const float* be3 = (const float*)d_in[16];
  const float* m3  = (const float*)d_in[17];
  const float* v3  = (const float*)d_in[18];
  const float* wf  = (const float*)d_in[19];
  const float* bf  = (const float*)d_in[20];
  const float* g4  = (const float*)d_in[21];
  const float* be4 = (const float*)d_in[22];
  const float* m4  = (const float*)d_in[23];
  const float* v4  = (const float*)d_in[24];
  const float* wl  = (const float*)d_in[25];
  const float* bl  = (const float*)d_in[26];
  const float* sa1 = (const float*)d_in[27];
  const float* sa2 = (const float*)d_in[28];
  const float* sa3 = (const float*)d_in[29];
  float* out = (float*)d_out;

  char* ws = (char*)d_ws;
  float* scal  = (float*)ws;
  float* h3acc = (float*)(ws + 1024);
  signed char* w2s = (signed char*)(ws + (512 << 10));
  signed char* w3s = (signed char*)(ws + (512 << 10) + 16384);
  _Float16* h1f = (_Float16*)(ws + (1L << 20));
  _Float16* h2f = (_Float16*)(ws + (1L << 20) + 512L * 32 * 2048 * 2);

  bnn_prep_kernel<<<1, 256, 0, stream>>>(w2, w3, wf, scal, h3acc, w2s, w3s);

  {
    const long total = 512L * 32 * 2048;
    const int blocks = (int)((total + 255) / 256);
    bnn_first_conv_kernel<<<blocks, 256, 0, stream>>>(x, w1, b1, g1, be1, m1,
                                                      v1, h1f);
  }

  bnn_bconv1_kernel<<<512 * 16, 128, 0, stream>>>(h1f, w2s, b2, g2, be2, m2,
                                                  v2, sa1, scal, h2f);

  bnn_bconv2_kernel<<<512 * 8, 256, 0, stream>>>(h2f, w3s, b3, g3, be3, m3, v3,
                                                 sa2, scal, h3acc);

  bnn_head_kernel<<<512, 64, 0, stream>>>(h3acc, wf, bf, g4, be4, m4, v4, wl,
                                          bl, sa3, scal, out);
}
